// EmoGene_49684181680531
// MI455X (gfx1250) — matrix operand order repaired and checked by reading, not compiled, not run on hardware
//
#include <hip/hip_runtime.h>
#include <hip/hip_bf16.h>
#include <math.h>

// Problem constants (match reference)
#define DD    64
#define SS    156
#define TGTD  274
#define BB    2048
#define VOCAB 30522

typedef __attribute__((ext_vector_type(16))) _Float16 v16h;
typedef __attribute__((ext_vector_type(8)))  _Float16 v8h;
typedef __attribute__((ext_vector_type(8)))  float    v8f;

// ---------------------------------------------------------------------------
// All-f16 WMMA GEMM:  C = act( alpha * (A[M,K] @ B[K,N]) + bias[N] )
//   A   : f16 row-major (M x K), optional row gather via gidx
//   Bt  : f16 "n-major" (Bt[n*ldbt + k] == B[k][n]) -> contiguous-k loads
//   Out : any non-null of { C (f32 row-major), C16 (f16 row-major),
//                           C16T (f16 transposed, C16T[n*M + m]) }
// Requirements: M%16==0, N%64==0, K%32==0; one wave owns a 16x64 C tile.
// K loop is software-pipelined (double-buffered): block k+1's loads are in
// flight while block k's 4 WMMAs execute, so WMMAs wait on partial loadcnt.
// ---------------------------------------------------------------------------
__global__ __launch_bounds__(128)
void gemm_f16_kernel(const _Float16* __restrict__ A, int lda,
                     const int* __restrict__ gidx,
                     const _Float16* __restrict__ Bt, int ldbt,
                     const float* __restrict__ bias,
                     float* __restrict__ C,
                     _Float16* __restrict__ C16,
                     _Float16* __restrict__ C16T, int ldc,
                     int M, int N, int K, float alpha, int act)
{
  const int lane = threadIdx.x & 31;
  const int wave = threadIdx.x >> 5;
  const int tilesN = N >> 6;
  const long long tile = (long long)blockIdx.x * 4 + wave;
  if (tile >= (long long)(M >> 4) * tilesN) return;   // wave-uniform

  const int tM = (int)(tile / tilesN);
  const int tN = (int)(tile % tilesN);
  const int row0 = tM << 4;
  const int col0 = tN << 6;
  const int hsel = lane >> 4;
  const int lm   = lane & 15;

  const long long asrc = gidx ? (long long)gidx[row0 + lm] : (long long)(row0 + lm);
  const _Float16* __restrict__ ap0 = A + asrc * lda + (hsel << 4);
  const _Float16* __restrict__ ap1 = ap0 + 8;
  const _Float16* __restrict__ b0p = Bt + (long long)(col0 + lm) * ldbt + (hsel << 4);
  const long long bs1 = (long long)16 * ldbt;

  v8f acc0 = {}, acc1 = {}, acc2 = {}, acc3 = {};

  // prologue: block 0 fragments
  v8h  al = *(const v8h*)(ap0);
  v8h  ah = *(const v8h*)(ap1);
  v16h b0 = *(const v16h*)(b0p);
  v16h b1 = *(const v16h*)(b0p + bs1);
  v16h b2 = *(const v16h*)(b0p + 2 * bs1);
  v16h b3 = *(const v16h*)(b0p + 3 * bs1);

  int k0 = 0;
  for (;;) {
    const int kn = k0 + 32;
    v16h a;
#pragma unroll
    for (int i = 0; i < 8; i++) { a[i] = al[i]; a[i + 8] = ah[i]; }

    if (kn < K) {                      // steady state: prefetch block kn
      __builtin_prefetch(ap0 + kn + 96, 0, 1);
      v8h  nal = *(const v8h*)(ap0 + kn);
      v8h  nah = *(const v8h*)(ap1 + kn);
      v16h nb0 = *(const v16h*)(b0p + kn);
      v16h nb1 = *(const v16h*)(b0p + bs1 + kn);
      v16h nb2 = *(const v16h*)(b0p + 2 * bs1 + kn);
      v16h nb3 = *(const v16h*)(b0p + 3 * bs1 + kn);

      acc0 = __builtin_amdgcn_wmma_f32_16x16x32_f16(false, a, false, b0, (short)0, acc0, false, false);
      acc1 = __builtin_amdgcn_wmma_f32_16x16x32_f16(false, a, false, b1, (short)0, acc1, false, false);
      acc2 = __builtin_amdgcn_wmma_f32_16x16x32_f16(false, a, false, b2, (short)0, acc2, false, false);
      acc3 = __builtin_amdgcn_wmma_f32_16x16x32_f16(false, a, false, b3, (short)0, acc3, false, false);

      al = nal; ah = nah; b0 = nb0; b1 = nb1; b2 = nb2; b3 = nb3;
      k0 = kn;
    } else {                           // epilogue block: no further loads
      acc0 = __builtin_amdgcn_wmma_f32_16x16x32_f16(false, a, false, b0, (short)0, acc0, false, false);
      acc1 = __builtin_amdgcn_wmma_f32_16x16x32_f16(false, a, false, b1, (short)0, acc1, false, false);
      acc2 = __builtin_amdgcn_wmma_f32_16x16x32_f16(false, a, false, b2, (short)0, acc2, false, false);
      acc3 = __builtin_amdgcn_wmma_f32_16x16x32_f16(false, a, false, b3, (short)0, acc3, false, false);
      break;
    }
  }

#pragma unroll
  for (int j = 0; j < 4; j++) {
    const v8f& acc = (j == 0) ? acc0 : (j == 1) ? acc1 : (j == 2) ? acc2 : acc3;
    const int col = col0 + j * 16 + lm;
    const float bi = bias ? bias[col] : 0.f;
    float o[8];
#pragma unroll
    for (int v = 0; v < 8; v++) {
      float t = acc[v] * alpha + bi;
      if (act == 1) t = 1.f / (1.f + __expf(-t));
      o[v] = t;
    }
    if (C) {
#pragma unroll
      for (int v = 0; v < 8; v++)
        C[(long long)(row0 + (hsel << 3) + v) * ldc + col] = o[v];
    }
    if (C16) {
#pragma unroll
      for (int v = 0; v < 8; v++)
        C16[(long long)(row0 + (hsel << 3) + v) * ldc + col] = (_Float16)o[v];
    }
    if (C16T) {                       // 8 consecutive rows -> one 16B store
      v8h o8;
#pragma unroll
      for (int v = 0; v < 8; v++) o8[v] = (_Float16)o[v];
      *(v8h*)(C16T + (long long)col * M + row0 + (hsel << 3)) = o8;
    }
  }
}

// ---------------------------------------------------------------------------
// f32-input WMMA GEMM with ragged-K tail (only for tgt @ Wtgt, K = 274).
// ---------------------------------------------------------------------------
__device__ __forceinline__ void ld8_cvt(const float* __restrict__ p,
                                        v16h& d, int base) {
  const float2* q = (const float2*)p;
#pragma unroll
  for (int i = 0; i < 4; i++) {
    float2 t = q[i];
    d[base + 2 * i]     = (_Float16)t.x;
    d[base + 2 * i + 1] = (_Float16)t.y;
  }
}

__global__ __launch_bounds__(128)
void gemm_f32_ragk_kernel(const float* __restrict__ A, int lda,
                          const float* __restrict__ Bm, int ldb,
                          const float* __restrict__ bias,
                          float* __restrict__ C, int ldc,
                          int M, int N, int K, int act)
{
  const int lane = threadIdx.x & 31;
  const int wave = threadIdx.x >> 5;
  const int tilesN = N >> 6;
  const long long tile = (long long)blockIdx.x * 4 + wave;
  if (tile >= (long long)(M >> 4) * tilesN) return;

  const int tM = (int)(tile / tilesN);
  const int tN = (int)(tile % tilesN);
  const int row0 = tM << 4, col0 = tN << 6;
  const int hsel = lane >> 4, lm = lane & 15;
  const float* __restrict__ Arow = A + (long long)(row0 + lm) * lda;

  v8f acc0 = {}, acc1 = {}, acc2 = {}, acc3 = {};
  const int K32 = K & ~31;

  for (int k0 = 0; k0 < K32; k0 += 32) {
    v16h a;
    ld8_cvt(Arow + k0 + (hsel << 4), a, 0);
    ld8_cvt(Arow + k0 + 8 + (hsel << 4), a, 8);
    v16h b0, b1, b2, b3;
    const float* brow = Bm + (long long)(k0 + (hsel << 4)) * ldb + col0 + lm;
#pragma unroll
    for (int e = 0; e < 16; e++) {
      const float* p = brow + (long long)e * ldb;
      b0[e] = (_Float16)p[0];  b1[e] = (_Float16)p[16];
      b2[e] = (_Float16)p[32]; b3[e] = (_Float16)p[48];
    }
    acc0 = __builtin_amdgcn_wmma_f32_16x16x32_f16(false, a, false, b0, (short)0, acc0, false, false);
    acc1 = __builtin_amdgcn_wmma_f32_16x16x32_f16(false, a, false, b1, (short)0, acc1, false, false);
    acc2 = __builtin_amdgcn_wmma_f32_16x16x32_f16(false, a, false, b2, (short)0, acc2, false, false);
    acc3 = __builtin_amdgcn_wmma_f32_16x16x32_f16(false, a, false, b3, (short)0, acc3, false, false);
  }
  if (K32 < K) {
    v16h a, b0, b1, b2, b3;
#pragma unroll
    for (int e = 0; e < 16; e++) {
      const int ka = K32 + (hsel << 4) + e;
      a[e] = (_Float16)((ka < K) ? Arow[ka] : 0.f);
      const int kb = K32 + (hsel << 4) + e;
      float v0 = 0.f, v1 = 0.f, v2 = 0.f, v3 = 0.f;
      if (kb < K) {
        const float* p = Bm + (long long)kb * ldb + col0 + lm;
        v0 = p[0]; v1 = p[16]; v2 = p[32]; v3 = p[48];
      }
      b0[e] = (_Float16)v0; b1[e] = (_Float16)v1;
      b2[e] = (_Float16)v2; b3[e] = (_Float16)v3;
    }
    acc0 = __builtin_amdgcn_wmma_f32_16x16x32_f16(false, a, false, b0, (short)0, acc0, false, false);
    acc1 = __builtin_amdgcn_wmma_f32_16x16x32_f16(false, a, false, b1, (short)0, acc1, false, false);
    acc2 = __builtin_amdgcn_wmma_f32_16x16x32_f16(false, a, false, b2, (short)0, acc2, false, false);
    acc3 = __builtin_amdgcn_wmma_f32_16x16x32_f16(false, a, false, b3, (short)0, acc3, false, false);
  }
#pragma unroll
  for (int j = 0; j < 4; j++) {
    const v8f& acc = (j == 0) ? acc0 : (j == 1) ? acc1 : (j == 2) ? acc2 : acc3;
    const int col = col0 + j * 16 + lm;
    const float bi = bias ? bias[col] : 0.f;
#pragma unroll
    for (int v = 0; v < 8; v++) {
      float o = acc[v] + bi;
      if (act == 1) o = 1.f / (1.f + __expf(-o));
      C[(long long)(row0 + (hsel << 3) + v) * ldc + col] = o;
    }
  }
}

// ---------------------------------------------------------------------------
// Conversion / elementwise / reduction kernels
// ---------------------------------------------------------------------------
__global__ void conv16_kernel(const float* __restrict__ s,
                              _Float16* __restrict__ d, int n) {
  int i = blockIdx.x * 256 + threadIdx.x;
  if (i < n) d[i] = (_Float16)s[i];
}

// W (Kd x Nd, f32) -> Wt (Nd x Kd, f16), Wt[n*Kd + k] = W[k*Nd + n]
__global__ void convT_kernel(const float* __restrict__ w,
                             _Float16* __restrict__ wt, int Kd, int Nd) {
  int i = blockIdx.x * 256 + threadIdx.x;
  if (i >= Kd * Nd) return;
  int k = i / Nd, n = i % Nd;
  wt[n * Kd + k] = (_Float16)w[i];
}

__global__ void zero_kernel(float* p, int n) {
  int i = blockIdx.x * 256 + threadIdx.x;
  if (i < n) p[i] = 0.f;
}

__global__ void bias_combine_kernel(const float* a0, const float* a1,
                                    const float* b0, const float* b1,
                                    float* oa, float* ob) {
  int i = threadIdx.x;
  if (i < DD) { oa[i] = a0[i] + a1[i]; ob[i] = b0[i] + b1[i]; }
}

// cat16[b,0:64] = x16[b*S+t,:] ; cat16[b,64:128] = cvt(h[b,:])
__global__ void build_cat_kernel(const _Float16* __restrict__ x16,
                                 const float* __restrict__ h,
                                 _Float16* __restrict__ cat16, int t) {
  int i = blockIdx.x * 256 + threadIdx.x;
  if (i >= BB * 2 * DD) return;
  int b = i >> 7, d = i & 127;
  cat16[i] = (d < DD) ? x16[((long long)b * SS + t) * DD + d]
                      : (_Float16)h[b * DD + (d - DD)];
}

// ch = ft * c (f32 + f16 copies)
__global__ void mul2_kernel(const float* a, const float* b,
                            float* o, _Float16* o16, int n) {
  int i = blockIdx.x * 256 + threadIdx.x;
  if (i < n) { float v = a[i] * b[i]; o[i] = v; o16[i] = (_Float16)v; }
}

// h = ao ; c = ch + ao ; e *= ut
__global__ void update_kernel(const float* __restrict__ ao,
                              const float* __restrict__ ch,
                              const float* __restrict__ ut,
                              float* h, float* c, float* e, int n) {
  int i = blockIdx.x * 256 + threadIdx.x;
  if (i < n) { h[i] = ao[i]; c[i] = ch[i] + ao[i]; e[i] *= ut[i]; }
}

// Row softmax of f32 scores, emit normalized probabilities in f16.
__global__ void softmax_rows_kernel(const float* __restrict__ S,
                                    _Float16* __restrict__ P, int n) {
  const float* r = S + (long long)blockIdx.x * n;
  _Float16* pr = P + (long long)blockIdx.x * n;
  __shared__ float red[256];
  int tid = threadIdx.x;
  float m = -1e30f;
  for (int i = tid; i < n; i += 256) m = fmaxf(m, r[i]);
  red[tid] = m; __syncthreads();
  for (int s = 128; s > 0; s >>= 1) {
    if (tid < s) red[tid] = fmaxf(red[tid], red[tid + s]);
    __syncthreads();
  }
  m = red[0]; __syncthreads();
  float sum = 0.f;
  for (int i = tid; i < n; i += 256) sum += __expf(r[i] - m);
  red[tid] = sum; __syncthreads();
  for (int s = 128; s > 0; s >>= 1) {
    if (tid < s) red[tid] += red[tid + s];
    __syncthreads();
  }
  float inv = 1.f / red[0];
  for (int i = tid; i < n; i += 256)
    pr[i] = (_Float16)(__expf(r[i] - m) * inv);
}

// vpre16 = cvt_f16( vpre_f32 + e[b] + pos[s] )
__global__ void add_pos_e_kernel(const float* __restrict__ vpre,
                                 const float* __restrict__ e,
                                 const float* __restrict__ pos,
                                 _Float16* __restrict__ out16) {
  long long i = (long long)blockIdx.x * 256 + threadIdx.x;
  if (i >= (long long)BB * SS * DD) return;
  int d = (int)(i % DD);
  long long bs = i / DD;
  int s = (int)(bs % SS);
  int b = (int)(bs / SS);
  out16[i] = (_Float16)(vpre[i] + e[b * DD + d] + pos[s * DD + d]);
}

// Tiny 2-head MHA, single query per batch, S=156 keys, Dh=32 (f16 K/V).
__global__ __launch_bounds__(128)
void mha_kernel(const float* __restrict__ qh,        // (B, 64) f32
                const _Float16* __restrict__ kh,     // (B, S, 64) f16
                const _Float16* __restrict__ vh,     // (B, S, 64) f16
                _Float16* __restrict__ out) {        // (B, 64) f16
  const int b = blockIdx.x;
  __shared__ float p[2][SS];
  __shared__ float qs[DD];
  __shared__ float inv[2];
  const int tid = threadIdx.x;
  if (tid < DD) qs[tid] = qh[b * DD + tid];
  __syncthreads();
  const float scale = 0.17677669529663687f;   // 1/sqrt(32)
  for (int idx = tid; idx < 2 * SS; idx += 128) {
    int hh = idx / SS, i = idx % SS;
    const _Float16* kr = kh + ((long long)b * SS + i) * DD + hh * 32;
    float s = 0.f;
#pragma unroll
    for (int d = 0; d < 32; d++) s += qs[hh * 32 + d] * (float)kr[d];
    p[hh][i] = s * scale;
  }
  __syncthreads();
  if (tid < 2) {
    float m = -1e30f;
    for (int i = 0; i < SS; i++) m = fmaxf(m, p[tid][i]);
    float sum = 0.f;
    for (int i = 0; i < SS; i++) {
      float ex = __expf(p[tid][i] - m); p[tid][i] = ex; sum += ex;
    }
    inv[tid] = 1.f / sum;
  }
  __syncthreads();
  for (int d = tid; d < DD; d += 128) {
    int hh = d >> 5;
    float o = 0.f;
    for (int i = 0; i < SS; i++)
      o += p[hh][i] * (float)vh[((long long)b * SS + i) * DD + d];
    out[b * DD + d] = (_Float16)(o * inv[hh]);
  }
}

// ---------------------------------------------------------------------------
// Host side
// ---------------------------------------------------------------------------
static inline void g16(const _Float16* A, int lda, const int* gidx,
                       const _Float16* Bt, int ldbt, const float* bias,
                       float* C, _Float16* C16, _Float16* C16T, int ldc,
                       int M, int N, int K, float alpha, int act,
                       hipStream_t stream) {
  long long tiles = (long long)(M >> 4) * (N >> 6);
  int blocks = (int)((tiles + 3) / 4);
  gemm_f16_kernel<<<blocks, 128, 0, stream>>>(A, lda, gidx, Bt, ldbt, bias,
                                              C, C16, C16T, ldc, M, N, K, alpha, act);
}

extern "C" void kernel_launch(void* const* d_in, const int* in_sizes, int n_in,
                              void* d_out, int out_size, void* d_ws, size_t ws_size,
                              hipStream_t stream) {
  (void)in_sizes; (void)n_in; (void)out_size; (void)ws_size;

  const int*   src_ids = (const int*)d_in[0];
  const float* tgt     = (const float*)d_in[1];
  const float* emb     = (const float*)d_in[2];
  const float* pos     = (const float*)d_in[3];
  const float* eWf     = (const float*)d_in[4];
  const float* ebf_lin = (const float*)d_in[5];
  const float* ebf     = (const float*)d_in[6];
  const float* eWq     = (const float*)d_in[7];
  const float* ebq     = (const float*)d_in[8];
  const float* eWk     = (const float*)d_in[9];
  const float* ebk     = (const float*)d_in[10];
  const float* eWv     = (const float*)d_in[11];
  const float* ebv     = (const float*)d_in[12];
  const float* eWu     = (const float*)d_in[13];
  const float* ebu_lin = (const float*)d_in[14];
  const float* ebu     = (const float*)d_in[15];
  const float* Wq_t    = (const float*)d_in[16];
  const float* bq_t    = (const float*)d_in[17];
  const float* Wk_t    = (const float*)d_in[18];
  const float* bk_t    = (const float*)d_in[19];
  const float* Wv_t    = (const float*)d_in[20];
  const float* bv_t    = (const float*)d_in[21];
  const float* Wsrc    = (const float*)d_in[22];
  const float* bsrc    = (const float*)d_in[23];
  const float* Wtgt    = (const float*)d_in[24];
  const float* btgt    = (const float*)d_in[25];
  const float* mWq     = (const float*)d_in[26];
  const float* mbq     = (const float*)d_in[27];
  const float* mWk     = (const float*)d_in[28];
  const float* mbk     = (const float*)d_in[29];
  const float* mWv     = (const float*)d_in[30];
  const float* mbv     = (const float*)d_in[31];
  const float* mWo     = (const float*)d_in[32];
  const float* mbo     = (const float*)d_in[33];

  // --- workspace carve-up (256B aligned) ---
  size_t off = 0;
  auto carveB = [&](size_t bytes) -> void* {
    void* p = (void*)((char*)d_ws + off);
    off += bytes;
    off = (off + 255) & ~(size_t)255;
    return p;
  };
  auto carveF = [&](size_t nf) -> float*    { return (float*)carveB(nf * 4); };
  auto carveH = [&](size_t nh) -> _Float16* { return (_Float16*)carveB(nh * 2); };

  const size_t BSD = (size_t)BB * SS * DD;
  float*    bigF  = carveF(BSD);              // vpre (f32, from ragged GEMM)
  float*    Sc    = carveF((size_t)BB * BB);  // attention scores (f32)
  _Float16* poolA = carveH(BSD);              // x16 -> k16 -> vhp16
  _Float16* poolB = carveH(BSD);              // vpre16 -> khp16
  _Float16* poolC = carveH(BSD);              // v16
  _Float16* P16   = carveH((size_t)BB * BB);  // softmax probs (f16)
  _Float16* emb16 = carveH((size_t)VOCAB * DD);
  // small f16
  _Float16* cat16 = carveH(BB * 2 * DD);
  _Float16* ch16  = carveH(BB * DD);
  _Float16* Q16   = carveH(BB * DD);
  _Float16* K16   = carveH(BB * DD);
  _Float16* V16T  = carveH(BB * DD);          // transposed (DD x BB)
  _Float16* ao16  = carveH(BB * DD);
  _Float16* h16   = carveH(BB * DD);
  _Float16* qv16  = carveH(BB * DD);
  _Float16* oh16  = carveH(BB * DD);
  // f16 transposed weights  (Wt[n*K + k])
  _Float16* eWfT  = carveH(128 * 64);
  _Float16* eWkT  = carveH(128 * 64);
  _Float16* eWvT  = carveH(128 * 64);
  _Float16* eWqT  = carveH(64 * 64);
  _Float16* eWuT  = carveH(64 * 64);
  _Float16* WsrcT = carveH(64 * 64);
  _Float16* WqtT  = carveH(64 * 64);
  _Float16* WktT  = carveH(64 * 64);
  _Float16* WvtT  = carveH(64 * 64);
  _Float16* mWqT  = carveH(64 * 64);
  _Float16* mWkT  = carveH(64 * 64);
  _Float16* mWvT  = carveH(64 * 64);
  _Float16* mWoT  = carveH(64 * 64);
  // small f32
  float* h  = carveF(BB * DD);
  float* c  = carveF(BB * DD);
  float* e  = carveF(BB * DD);
  float* ch = carveF(BB * DD);
  float* ft = carveF(BB * DD);
  float* ao = carveF(BB * DD);
  float* ut = carveF(BB * DD);
  float* qhb = carveF(BB * DD);
  float* biasF = carveF(DD);
  float* biasU = carveF(DD);

  const int nBD = BB * DD;
  const int gBD = (nBD + 255) / 256;
  const int g64x64 = (64 * 64 + 255) / 256;
  const int g128x64 = (128 * 64 + 255) / 256;

  // --- one-time conversions ---
  zero_kernel<<<gBD, 256, 0, stream>>>(h, nBD);
  zero_kernel<<<gBD, 256, 0, stream>>>(c, nBD);
  zero_kernel<<<gBD, 256, 0, stream>>>(e, nBD);
  bias_combine_kernel<<<1, 64, 0, stream>>>(ebf_lin, ebf, ebu_lin, ebu, biasF, biasU);
  conv16_kernel<<<(VOCAB * DD + 255) / 256, 256, 0, stream>>>(emb, emb16, VOCAB * DD);
  convT_kernel<<<g128x64, 256, 0, stream>>>(eWf, eWfT, 128, 64);
  convT_kernel<<<g128x64, 256, 0, stream>>>(eWk, eWkT, 128, 64);
  convT_kernel<<<g128x64, 256, 0, stream>>>(eWv, eWvT, 128, 64);
  convT_kernel<<<g64x64, 256, 0, stream>>>(eWq, eWqT, 64, 64);
  convT_kernel<<<g64x64, 256, 0, stream>>>(eWu, eWuT, 64, 64);
  convT_kernel<<<g64x64, 256, 0, stream>>>(Wsrc, WsrcT, 64, 64);
  convT_kernel<<<g64x64, 256, 0, stream>>>(Wq_t, WqtT, 64, 64);
  convT_kernel<<<g64x64, 256, 0, stream>>>(Wk_t, WktT, 64, 64);
  convT_kernel<<<g64x64, 256, 0, stream>>>(Wv_t, WvtT, 64, 64);
  convT_kernel<<<g64x64, 256, 0, stream>>>(mWq, mWqT, 64, 64);
  convT_kernel<<<g64x64, 256, 0, stream>>>(mWk, mWkT, 64, 64);
  convT_kernel<<<g64x64, 256, 0, stream>>>(mWv, mWvT, 64, 64);
  convT_kernel<<<g64x64, 256, 0, stream>>>(mWo, mWoT, 64, 64);

  // x16 = f16( emb[src_ids] @ Wsrc + bsrc )   (gathered-A, M = B*S)
  g16(emb16, DD, src_ids, WsrcT, DD, bsrc, nullptr, poolA, nullptr, DD,
      BB * SS, DD, DD, 1.f, 0, stream);

  // ---- recurrent scan: 156 sequential steps ----
  const int gCat = (BB * 2 * DD + 255) / 256;
  for (int t = 0; t < SS; t++) {
    build_cat_kernel<<<gCat, 256, 0, stream>>>(poolA, h, cat16, t);
    // ft = sigmoid(cat @ eWf + biasF)
    g16(cat16, 2 * DD, nullptr, eWfT, 2 * DD, biasF, ft, nullptr, nullptr, DD,
        BB, DD, 2 * DD, 1.f, 1, stream);
    mul2_kernel<<<gBD, 256, 0, stream>>>(ft, c, ch, ch16, nBD);
    // Q/K/V projections (f16 outputs; V transposed for the P@V GEMM)
    g16(ch16, DD, nullptr, eWqT, DD, ebq, nullptr, Q16, nullptr, DD,
        BB, DD, DD, 1.f, 0, stream);
    g16(cat16, 2 * DD, nullptr, eWkT, 2 * DD, ebk, nullptr, K16, nullptr, DD,
        BB, DD, 2 * DD, 1.f, 0, stream);
    g16(cat16, 2 * DD, nullptr, eWvT, 2 * DD, ebv, nullptr, nullptr, V16T, DD,
        BB, DD, 2 * DD, 1.f, 0, stream);
    // Sc = (Q @ K^T)/8 : Bt = K16 row-major is exactly n-major here
    g16(Q16, DD, nullptr, K16, DD, nullptr, Sc, nullptr, nullptr, BB,
        BB, BB, DD, 0.125f, 0, stream);
    softmax_rows_kernel<<<BB, 256, 0, stream>>>(Sc, P16, BB);
    // ao = P @ V : Bt = V16T (n-major, ld = BB); f32 + f16 outputs
    g16(P16, BB, nullptr, V16T, BB, nullptr, ao, ao16, nullptr, DD,
        BB, DD, BB, 1.f, 0, stream);
    // ut = sigmoid(ao @ eWu + biasU)
    g16(ao16, DD, nullptr, eWuT, DD, biasU, ut, nullptr, nullptr, DD,
        BB, DD, DD, 1.f, 1, stream);
    update_kernel<<<gBD, 256, 0, stream>>>(ao, ch, ut, h, c, e, nBD);
  }

  // ---- top-level projections + 2-head MHA ----
  conv16_kernel<<<gBD, 256, 0, stream>>>(h, h16, nBD);
  // qvec = h @ Wq_t + bq_t  (f16)
  g16(h16, DD, nullptr, WqtT, DD, bq_t, nullptr, qv16, nullptr, DD,
      BB, DD, DD, 1.f, 0, stream);
  // vpre = tgt @ Wtgt + btgt  (f32-input ragged-K GEMM)
  {
    long long tiles = (long long)((BB * SS) >> 4) * (DD >> 6);
    int blocks = (int)((tiles + 3) / 4);
    gemm_f32_ragk_kernel<<<blocks, 128, 0, stream>>>(tgt, TGTD, Wtgt, DD, btgt,
                                                     bigF, DD, BB * SS, DD, TGTD, 0);
  }
  // vpre16 = f16(vpre + e[b] + pos[s])     -> poolB
  {
    long long n = (long long)BB * SS * DD;
    add_pos_e_kernel<<<(int)((n + 255) / 256), 256, 0, stream>>>(bigF, e, pos, poolB);
  }
  // k = vpre @ Wk_t + bk_t (f16, poolA reusable after scan)
  g16(poolB, DD, nullptr, WktT, DD, bk_t, nullptr, poolA, nullptr, DD,
      BB * SS, DD, DD, 1.f, 0, stream);
  // v = vpre @ Wv_t + bv_t (f16, poolC)
  g16(poolB, DD, nullptr, WvtT, DD, bv_t, nullptr, poolC, nullptr, DD,
      BB * SS, DD, DD, 1.f, 0, stream);
  // MHA in-projections
  g16(qv16, DD, nullptr, mWqT, DD, mbq, qhb, nullptr, nullptr, DD,
      BB, DD, DD, 1.f, 0, stream);
  g16(poolA, DD, nullptr, mWkT, DD, mbk, nullptr, poolB, nullptr, DD,   // khp16
      BB * SS, DD, DD, 1.f, 0, stream);
  g16(poolC, DD, nullptr, mWvT, DD, mbv, nullptr, poolA, nullptr, DD,   // vhp16
      BB * SS, DD, DD, 1.f, 0, stream);
  mha_kernel<<<BB, 128, 0, stream>>>(qhb, poolB, poolA, oh16);
  // out = o @ mWo + mbo  (f32, straight to d_out)
  g16(oh16, DD, nullptr, mWoT, DD, mbo, (float*)d_out, nullptr, nullptr, DD,
      BB, DD, DD, 1.f, 0, stream);
}
